// Hgru2_16767552323886
// MI455X (gfx1250) — matrix operand order repaired and checked by reading, not compiled, not run on hardware
//
#include <hip/hip_runtime.h>
#include <hip/hip_bf16.h>

// ---------------- problem constants ----------------
constexpr int Bb = 4;
constexpr int Nn = 4096;
constexpr int Dd = 2048;
constexpr int Ee = 128;
constexpr int Hh = Dd / Ee;   // 16
constexpr int CH = 64;        // chunk
constexpr int NC = Nn / CH;   // 64 chunks
constexpr int Mrows = Bb * Nn; // 16384

// ---------------- vector types ----------------
typedef __attribute__((ext_vector_type(8)))  __bf16 v8bf;
typedef __attribute__((ext_vector_type(16))) __bf16 v16bf;
typedef __attribute__((ext_vector_type(8)))  float  v8f;
typedef __attribute__((ext_vector_type(4)))  unsigned int u32x4;
typedef __attribute__((ext_vector_type(8)))  int i32x8;
typedef __attribute__((ext_vector_type(4)))  int i32x4;

#define AS_LOCAL __attribute__((address_space(3)))

__device__ __forceinline__ unsigned lds_byte_addr(const void* p) {
  // explicit generic->AS3 cast, then ptrtoint: AS3 pointers are LDS byte offsets
  return (unsigned)(unsigned long long)(AS_LOCAL const void*)p;
}

// ---------------- CDNA5 async global->LDS copy (ASYNCcnt path) ----------------
// Emitted via inline asm (ISA §15.18): VDST = LDS byte address VGPR,
// VADDR = 64-bit global address, no SADDR.
__device__ __forceinline__ void copy16_to_lds(const __bf16* g, __bf16* l) {
  const unsigned lds = lds_byte_addr(l);
  const unsigned long long ga = (unsigned long long)g;
  asm volatile("global_load_async_to_lds_b128 %0, %1, off"
               :
               : "v"(lds), "v"(ga)
               : "memory");
}

__device__ __forceinline__ void wait_async_copies() {
  __builtin_amdgcn_s_wait_asynccnt(0);
}

// ---------------- CDNA5 Tensor Data Mover (TENSORcnt path) ----------------
// D# descriptor per ISA §8.3-8.6: 2-D tile of `tile_d1` lines x `tile_d0`
// elements from a row-major tensor (line stride d0_stride elements), with
// LDS line padding (pad_interval_code: 2^(c+1) dwords, pad_amount_code: c+1
// dwords). Issued once per wave; completion tracked with TENSORcnt.
// This toolchain exposes the 6-arg builtin: (g0, g1, g2, g3, g4, cpol).
__device__ __forceinline__ void tdm_load_2d(
    unsigned lds_addr, const void* gptr,
    unsigned elem_log2,        // 0:1B 1:2B 2:4B 3:8B
    unsigned tile_d0, unsigned tile_d1,
    unsigned tensor_d0, unsigned d0_stride, unsigned tensor_d1,
    unsigned pad_interval_code, unsigned pad_amount_code) {
  const unsigned long long ga = (unsigned long long)gptr;
  u32x4 g0;
  g0[0] = 1u;                                              // count=1, user D#
  g0[1] = lds_addr;                                        // lds_addr [63:32]
  g0[2] = (unsigned)(ga & 0xffffffffu);                    // global_addr[31:0]
  g0[3] = (unsigned)((ga >> 32) & 0x1ffffffu) | (2u << 30); // addr[56:32]|type=2
  i32x8 g1;
  g1[0] = (int)((elem_log2 << 16) | (1u << 20) |           // data_size|pad_en
                (pad_interval_code << 22) | (pad_amount_code << 25));
  g1[1] = (int)((tensor_d0 & 0xffffu) << 16);              // td0[15:0] @ [31:16]
  g1[2] = (int)(((tensor_d0 >> 16) & 0xffffu) | ((tensor_d1 & 0xffffu) << 16));
  g1[3] = (int)(((tensor_d1 >> 16) & 0xffffu) | ((tile_d0 & 0xffffu) << 16));
  g1[4] = (int)(tile_d1 & 0xffffu);                        // tile_dim1; dim2=0
  g1[5] = (int)d0_stride;                                  // td0_stride[31:0]
  g1[6] = 0;                                               // stride hi, d1_stride lo
  g1[7] = 0;
  i32x4 gz4 = {0, 0, 0, 0};                                // groups 2/3: 2-D only
  i32x8 gz8 = {0, 0, 0, 0, 0, 0, 0, 0};                    // extra group: unused
  __builtin_amdgcn_tensor_load_to_lds(g0, g1, gz4, gz4, gz8, 0);
}

__device__ __forceinline__ void wait_tensor_loads() {
  __builtin_amdgcn_s_wait_tensorcnt(0);   // must be a constant integer
}

__device__ __forceinline__ v16bf cat8(v8bf lo, v8bf hi) {
  return __builtin_shufflevector(lo, hi, 0,1,2,3,4,5,6,7,8,9,10,11,12,13,14,15);
}

// A-fragment (16x32, bf16) from an LDS tile stored row-major [m][k], row stride ld.
// ISA layout: lanes 0-15 hold M=lane, K 0-7 (v0-3) and 16-23 (v4-7);
//             lanes 16-31 hold M=lane-16, K 8-15 and 24-31.
__device__ __forceinline__ v16bf load_frag_a(const __bf16* base, int ld, int lane) {
  int half = lane >> 4;
  int m    = lane & 15;
  const __bf16* p = base + m * ld + half * 8;
  v8bf lo = *(const v8bf*)(p);
  v8bf hi = *(const v8bf*)(p + 16);
  return cat8(lo, hi);
}

// B-fragment (32x16, bf16): B[k][n]. LDS tile holds B transposed, i.e. [n][k]
// row-major with row stride ld. Lane n<16 reads K 0-7 and 16-23 of column n;
// lane n+16 reads K 8-15 and 24-31 (same order as the A-fragment).
__device__ __forceinline__ v16bf load_frag_b(const __bf16* base, int ld, int lane) {
  int n  = lane & 15;
  int kk = (lane >> 4) * 8;
  const __bf16* p = base + n * ld + kk;
  v8bf lo = *(const v8bf*)(p);
  v8bf hi = *(const v8bf*)(p + 16);
  return cat8(lo, hi);
}

__device__ __forceinline__ v8f wmma_bf16(v16bf a, v16bf b, v8f c) {
  return __builtin_amdgcn_wmma_f32_16x16x32_bf16(
      /*neg_a=*/false, a, /*neg_b=*/false, b,
      /*c_mod=*/(short)0, c, /*reuse_a=*/false, /*reuse_b=*/false);
}

// ---------------- f32 -> bf16 convert ----------------
__global__ void cvt_bf16_kernel(const float* __restrict__ in,
                                __bf16* __restrict__ out, size_t n) {
  size_t i = (size_t)blockIdx.x * blockDim.x + threadIdx.x;
  size_t stride = (size_t)gridDim.x * blockDim.x;
  for (; i < n; i += stride) out[i] = (__bf16)in[i];
}

// ---------------- tiled bf16 GEMM: async A staging + TDM W staging ----------------
// C[m,n] = sum_k A[m,k] * W[n,k]  (i.e. A @ W^T), M x N, K contraction.
// MODE 0: outf = y                (final projection, f32 out)
// MODE 1: outb = silu(y)          (q)
// MODE 2: f = sigmoid(y); outf = log_sigmoid(f); outb = sigmoid(-f)  (log_f, k)
// MODE 3: outb = y                (v)
constexpr int TM = 128, TN = 128, TK = 32, LDT = 40; // padded LDS stride

template <int MODE>
__global__ void gemm_bf16_kernel(const __bf16* __restrict__ A,
                                 const __bf16* __restrict__ W,
                                 __bf16* __restrict__ outb,
                                 float* __restrict__ outf,
                                 int M, int N, int K) {
  __shared__ __bf16 As[2][TM * LDT];
  __shared__ __bf16 Ws[2][TN * LDT];

  const int tid  = threadIdx.x;
  const int lane = tid & 31;
  const int wave = tid >> 5;          // 8 waves
  const int bm   = blockIdx.y * TM;
  const int bn   = blockIdx.x * TN;
  const int wm   = (wave >> 2) * 64;  // 2 wave rows x 64
  const int wn   = (wave & 3) * 32;   // 4 wave cols x 32

  v8f acc[4][2] = {};

  const int lr = tid >> 1;            // 0..127
  const int lc = (tid & 1) * 16;      // 0 or 16

  auto stageA = [&](int buf, int k0) {   // per-lane async engine
    const __bf16* ga = A + (size_t)(bm + lr) * K + k0 + lc;
    copy16_to_lds(ga,     As[buf] + lr * LDT + lc);
    copy16_to_lds(ga + 8, As[buf] + lr * LDT + lc + 8);
  };
  auto stageW = [&](int buf, int k0) {   // one TDM descriptor per tile
    if (wave == 0)
      tdm_load_2d(lds_byte_addr(Ws[buf]), W + (size_t)bn * K + k0,
                  /*elem=*/1, /*tile_d0=*/TK, /*tile_d1=*/TN,
                  /*tensor_d0=*/(unsigned)K, /*stride=*/(unsigned)K,
                  /*tensor_d1=*/TN,
                  /*pad: 16 dwords per line*/3, /*pad 4 dwords*/3);
  };

  stageA(0, 0);
  stageW(0, 0);

  int cur = 0;
  for (int k0 = 0; k0 < K; k0 += TK) {
    wait_async_copies();                   // A(cur) landed (per wave)
    if (wave == 0) wait_tensor_loads();    // W(cur) landed
    __syncthreads();                       // visible to all; prev readers done

    if (k0 + TK < K) {                     // kick next buffer under compute
      stageA(cur ^ 1, k0 + TK);
      stageW(cur ^ 1, k0 + TK);
    }

    v16bf bfrag[2];
#pragma unroll
    for (int nj = 0; nj < 2; ++nj)
      bfrag[nj] = load_frag_b(Ws[cur] + (wn + nj * 16) * LDT, LDT, lane);
#pragma unroll
    for (int mi = 0; mi < 4; ++mi) {
      v16bf afrag = load_frag_a(As[cur] + (wm + mi * 16) * LDT, LDT, lane);
#pragma unroll
      for (int nj = 0; nj < 2; ++nj)
        acc[mi][nj] = wmma_bf16(afrag, bfrag[nj], acc[mi][nj]);
    }
    cur ^= 1;
  }

  // epilogue: lane holds column n = lane&15; VGPR i holds row i + (lane>>4)*8
  const int nloc = lane & 15;
  const int mh   = (lane >> 4) * 8;
#pragma unroll
  for (int mi = 0; mi < 4; ++mi) {
#pragma unroll
    for (int nj = 0; nj < 2; ++nj) {
      const int col = bn + wn + nj * 16 + nloc;
#pragma unroll
      for (int i = 0; i < 8; ++i) {
        const int row = bm + wm + mi * 16 + mh + i;
        const size_t idx = (size_t)row * N + col;
        const float y = acc[mi][nj][i];
        if (MODE == 0) {
          outf[idx] = y;
        } else if (MODE == 1) {
          outb[idx] = (__bf16)(y / (1.0f + __expf(-y)));     // silu
        } else if (MODE == 2) {
          const float f = 1.0f / (1.0f + __expf(-y));        // sigmoid(y)
          outf[idx] = -log1pf(__expf(-f));                   // log_sigmoid(f)
          outb[idx] = (__bf16)(1.0f / (1.0f + __expf(f)));   // k = 1 - sigmoid(f)
        } else {
          outb[idx] = (__bf16)y;
        }
      }
    }
  }
}

// ---------------- GLA chunked scan ----------------
// One block per (b, h, f-half); FSPLIT trades +17% redundant score FLOPs for
// 2x WGP occupancy. Chunk inputs (q,k,v,log_f tiles) are DMA'd by the TDM and
// double-buffered across the sequential scan, hiding global latency behind
// the chunk's WMMAs. log_f cumsum is computed in place in the DMA tile.
constexpr int FSPLIT = 2;
constexpr int Fw     = Ee / FSPLIT;  // 64 local f columns

constexpr int LDA_  = 132;  // f32 stride: lf tile -> cumsum A (in place)
constexpr int LDQ_  = 136;  // bf16 [64][128] stride    qraw/kraw, qd, kd
constexpr int LDV_  = 72;   // bf16 [.][64] stride      vraw, vT, krT
constexpr int LDSC_ = 72;   // bf16 scores stride       [64][LDSC_]
constexpr int LDSF_ = 68;   // f32 S stride             [128][LDSF_] (Fw cols)
constexpr int LDST_ = 136;  // bf16 S^T stride          [Fw][LDST_]  (Ee cols)

constexpr size_t GLA_SMEM =
    (size_t)2 * 64 * LDQ_ * 2 * 2 +              // qraw, kraw (double buffered)
    (size_t)2 * 64 * LDV_ * 2 +                  // vraw (double buffered)
    (size_t)2 * 64 * LDA_ * 4 +                  // lf/A (double buffered)
    128 * 4 +                                    // decay
    (size_t)64 * LDQ_ * 2 * 2 +                  // qdec, kdec
    (size_t)Fw  * LDV_ * 2 +                     // v^T   [f][t]
    (size_t)128 * LDV_ * 2 +                     // krem^T[e][t]
    (size_t)64 * LDSC_ * 2 +                     // scores
    (size_t)128 * LDSF_ * 4 +                    // S f32  [e][f]
    (size_t)Fw  * LDST_ * 2;                     // S^T bf16 [f][e]  (= 280,064 B)

__global__ void gla_kernel(const __bf16* __restrict__ qg,
                           const __bf16* __restrict__ kg,
                           const __bf16* __restrict__ vg,
                           const float*  __restrict__ lfg,
                           float* __restrict__ og) {
  extern __shared__ char smem[];
  __bf16* qraw  = (__bf16*)smem;                  // [2][64*LDQ_]
  __bf16* kraw  = qraw + 2 * 64 * LDQ_;           // [2][64*LDQ_]
  __bf16* vraw  = kraw + 2 * 64 * LDQ_;           // [2][64*LDV_]
  float*  lfA   = (float*)(vraw + 2 * 64 * LDV_); // [2][64*LDA_] lf -> cumsum A
  float*  decay = lfA + 2 * 64 * LDA_;            // [128]
  __bf16* qd    = (__bf16*)(decay + 128);         // [64][LDQ_]
  __bf16* kd    = qd + 64 * LDQ_;                 // [64][LDQ_]
  __bf16* vT    = kd + 64 * LDQ_;                 // [Fw][LDV_]   v^T[f][t]
  __bf16* krT   = vT + Fw * LDV_;                 // [128][LDV_]  krem^T[e][t]
  __bf16* sc    = krT + 128 * LDV_;               // [64][LDSC_]
  float*  S     = (float*)(sc + 64 * LDSC_);      // [128][LDSF_]
  __bf16* ST    = (__bf16*)(S + 128 * LDSF_);     // [Fw][LDST_]  S^T[f][e]

  const int b     = blockIdx.x / (Hh * FSPLIT);
  const int rem   = blockIdx.x % (Hh * FSPLIT);
  const int h     = rem / FSPLIT;
  const int fs    = rem % FSPLIT;
  const int fbase = fs * Fw;
  const int tid   = threadIdx.x;
  const int lane  = tid & 31;
  const int wave  = tid >> 5;
  const size_t colbase = (size_t)h * Ee;

  auto tdm_chunk = [&](int c, int buf) {  // 4 TDM descriptors per chunk
    const size_t row0 = (size_t)b * Nn + (size_t)c * CH;
    const size_t g = row0 * Dd + colbase;
    tdm_load_2d(lds_byte_addr(qraw + buf * 64 * LDQ_), qg + g,
                1, Ee, CH, Dd, Dd, CH, /*64 dw*/5, 3);
    tdm_load_2d(lds_byte_addr(kraw + buf * 64 * LDQ_), kg + g,
                1, Ee, CH, Dd, Dd, CH, 5, 3);
    tdm_load_2d(lds_byte_addr(vraw + buf * 64 * LDV_), vg + g + fbase,
                1, Fw, CH, Dd, Dd, CH, /*32 dw*/4, 3);
    tdm_load_2d(lds_byte_addr(lfA + buf * 64 * LDA_), lfg + g,
                2, Ee, CH, Dd, Dd, CH, /*128 dw*/6, 3);
  };

  for (int i = tid; i < 128 * LDSF_; i += 256) S[i]  = 0.0f;
  for (int i = tid; i < Fw * LDST_;  i += 256) ST[i] = (__bf16)0.0f;

  if (wave == 0) tdm_chunk(0, 0);
  __syncthreads();

  int cur = 0;
  for (int c = 0; c < NC; ++c) {
    const size_t row0 = (size_t)b * Nn + (size_t)c * CH;

    if (wave == 0) wait_tensor_loads();   // chunk c tiles landed
    __syncthreads();                      // visible; prev-buffer readers done
    if (wave == 0 && c + 1 < NC) tdm_chunk(c + 1, cur ^ 1);  // prefetch next

    __bf16* qr = qraw + cur * 64 * LDQ_;
    __bf16* kr = kraw + cur * 64 * LDQ_;
    __bf16* vr = vraw + cur * 64 * LDV_;
    float*  Ac = lfA  + cur * 64 * LDA_;

    // ---- stage 1: per-channel cumsum of log_f, in place ----
    if (tid < Ee) {
      const int e = tid;
      float a = 0.0f;
      for (int t = 0; t < CH; ++t) {
        a += Ac[t * LDA_ + e];
        Ac[t * LDA_ + e] = a;
      }
      decay[e] = __expf(a);  // exp(A_last)
    }
    __syncthreads();

    // ---- stage 2a: decayed q/k fragments (full E) ----
    for (int idx = tid; idx < CH * Ee; idx += 256) {
      const int t = idx >> 7;          // /128
      const int e = idx & 127;
      const float a     = Ac[t * LDA_ + e];
      const float alast = Ac[63 * LDA_ + e];
      const float qv = (float)qr[t * LDQ_ + e];
      const float kv = (float)kr[t * LDQ_ + e];
      qd[t * LDQ_ + e]  = (__bf16)(qv * __expf(a));
      kd[t * LDQ_ + e]  = (__bf16)(kv * __expf(-a));
      krT[e * LDV_ + t] = (__bf16)(kv * __expf(alast - a));
    }
    // ---- stage 2b: v^T for this f-half (LDS -> LDS transpose) ----
    for (int idx = tid; idx < CH * Fw; idx += 256) {
      const int t = idx >> 6;          // /64
      const int f = idx & 63;
      vT[f * LDV_ + t] = vr[t * LDV_ + f];
    }
    __syncthreads();

    // ---- stage 3: scores = q_dec @ k_dec^T  (64x64, K=128), tril mask ----
    for (int ti = wave * 2; ti < wave * 2 + 2; ++ti) {
      const int mi = ti >> 2, nj = ti & 3;
      v8f a = {};
#pragma unroll
      for (int ke = 0; ke < 4; ++ke) {
        v16bf af = load_frag_a(qd + (mi * 16) * LDQ_ + ke * 32, LDQ_, lane);
        v16bf bf = load_frag_b(kd + (nj * 16) * LDQ_ + ke * 32, LDQ_, lane);
        a = wmma_bf16(af, bf, a);
      }
      const int nloc = lane & 15, mh = (lane >> 4) * 8;
#pragma unroll
      for (int i = 0; i < 8; ++i) {
        const int tt = mi * 16 + mh + i, ss = nj * 16 + nloc;
        sc[tt * LDSC_ + ss] = (__bf16)((tt >= ss) ? a[i] : 0.0f);
      }
    }
    __syncthreads();

    // ---- stage 4: o = scores @ v + q_dec @ S   (64 x Fw) ----
    for (int ti = wave * 2; ti < wave * 2 + 2; ++ti) {
      const int mi = ti >> 2, nj = ti & 3;   // 4 x 4 tiles of 16x16
      v8f a = {};
#pragma unroll
      for (int ks = 0; ks < 2; ++ks) {  // intra: K = 64 over s
        v16bf af = load_frag_a(sc + (mi * 16) * LDSC_ + ks * 32, LDSC_, lane);
        v16bf bf = load_frag_b(vT + (nj * 16) * LDV_  + ks * 32, LDV_,  lane);
        a = wmma_bf16(af, bf, a);
      }
#pragma unroll
      for (int ke = 0; ke < 4; ++ke) {  // inter: K = 128 over e
        v16bf af = load_frag_a(qd + (mi * 16) * LDQ_  + ke * 32, LDQ_,  lane);
        v16bf bf = load_frag_b(ST + (nj * 16) * LDST_ + ke * 32, LDST_, lane);
        a = wmma_bf16(af, bf, a);
      }
      const int nloc = lane & 15, mh = (lane >> 4) * 8;
#pragma unroll
      for (int i = 0; i < 8; ++i) {
        const int tt = mi * 16 + mh + i, ff = nj * 16 + nloc;
        og[(row0 + tt) * Dd + colbase + fbase + ff] = a[i];
      }
    }
    __syncthreads();

    // ---- stage 5: S = diag(exp(A_last)) S + krem^T @ v  (128 x Fw, K=64) ----
    {
      const int ei = wave;  // 8 waves x 16 state rows
#pragma unroll
      for (int fj = 0; fj < 4; ++fj) {
        v8f a = {};
#pragma unroll
        for (int kt = 0; kt < 2; ++kt) {
          v16bf af = load_frag_a(krT + (ei * 16) * LDV_ + kt * 32, LDV_, lane);
          v16bf bf = load_frag_b(vT  + (fj * 16) * LDV_ + kt * 32, LDV_, lane);
          a = wmma_bf16(af, bf, a);
        }
        const int nloc = lane & 15, mh = (lane >> 4) * 8;
#pragma unroll
        for (int i = 0; i < 8; ++i) {
          const int ee = ei * 16 + mh + i, ff = fj * 16 + nloc;
          const float sn = decay[ee] * S[ee * LDSF_ + ff] + a[i];
          S[ee * LDSF_ + ff] = sn;
          ST[ff * LDST_ + ee] = (__bf16)sn;
        }
      }
    }
    __syncthreads();
    cur ^= 1;
  }
}

// ---------------- RMSNorm (f32 in, bf16 out) ----------------
__global__ void rmsnorm_kernel(const float* __restrict__ og,
                               const float* __restrict__ w,
                               __bf16* __restrict__ onb) {
  __shared__ float red[256];
  const int row = blockIdx.x;
  const float* p = og + (size_t)row * Dd;
  float s = 0.0f;
  for (int j = threadIdx.x; j < Dd; j += 256) { const float v = p[j]; s += v * v; }
  red[threadIdx.x] = s;
  __syncthreads();
  for (int off = 128; off > 0; off >>= 1) {
    if (threadIdx.x < off) red[threadIdx.x] += red[threadIdx.x + off];
    __syncthreads();
  }
  const float scale = rsqrtf(red[0] / (float)Dd + 1e-6f);
  for (int j = threadIdx.x; j < Dd; j += 256)
    onb[(size_t)row * Dd + j] = (__bf16)(p[j] * scale * w[j]);
}

// ---------------- host launcher ----------------
extern "C" void kernel_launch(void* const* d_in, const int* in_sizes, int n_in,
                              void* d_out, int out_size, void* d_ws, size_t ws_size,
                              hipStream_t stream) {
  const float* x      = (const float*)d_in[0];
  const float* Wq     = (const float*)d_in[1];
  const float* Wk     = (const float*)d_in[2];
  const float* Wv     = (const float*)d_in[3];
  const float* Wo     = (const float*)d_in[4];
  const float* norm_w = (const float*)d_in[5];

  const size_t MD = (size_t)Mrows * Dd;   // 33.5M elements
  const size_t DD = (size_t)Dd * Dd;      // 4.2M elements

  char* ws = (char*)d_ws;
  size_t off = 0;
  auto carve = [&](size_t bytes) { char* p = ws + off; off += (bytes + 255) & ~(size_t)255; return p; };
  __bf16* xb  = (__bf16*)carve(MD * 2);
  __bf16* wqb = (__bf16*)carve(DD * 2);
  __bf16* wkb = (__bf16*)carve(DD * 2);
  __bf16* wvb = (__bf16*)carve(DD * 2);
  __bf16* wob = (__bf16*)carve(DD * 2);
  __bf16* qb  = (__bf16*)carve(MD * 2);
  __bf16* kb  = (__bf16*)carve(MD * 2);
  __bf16* vb  = (__bf16*)carve(MD * 2);
  float*  lf  = (float*)carve(MD * 4);
  float*  og  = (float*)carve(MD * 4);
  __bf16* onb = (__bf16*)carve(MD * 2);

  // 1) bf16 conversions
  cvt_bf16_kernel<<<1024, 256, 0, stream>>>(x,  xb,  MD);
  cvt_bf16_kernel<<<512,  256, 0, stream>>>(Wq, wqb, DD);
  cvt_bf16_kernel<<<512,  256, 0, stream>>>(Wk, wkb, DD);
  cvt_bf16_kernel<<<512,  256, 0, stream>>>(Wv, wvb, DD);
  cvt_bf16_kernel<<<512,  256, 0, stream>>>(Wo, wob, DD);

  // 2) projections (A @ W^T): WMMA bf16, async A staging + TDM W staging
  dim3 ggrid(Dd / TN, Mrows / TM);   // (16, 128)
  gemm_bf16_kernel<1><<<ggrid, 256, 0, stream>>>(xb, wqb, qb, nullptr, Mrows, Dd, Dd);
  gemm_bf16_kernel<2><<<ggrid, 256, 0, stream>>>(xb, wkb, kb, lf,      Mrows, Dd, Dd);
  gemm_bf16_kernel<3><<<ggrid, 256, 0, stream>>>(xb, wvb, vb, nullptr, Mrows, Dd, Dd);

  // 3) chunked GLA scan: one block per (b, h, f-half), TDM-pipelined chunks
  (void)hipFuncSetAttribute((const void*)gla_kernel,
                            hipFuncAttributeMaxDynamicSharedMemorySize, (int)GLA_SMEM);
  gla_kernel<<<Bb * Hh * FSPLIT, 256, GLA_SMEM, stream>>>(qb, kb, vb, lf, og);

  // 4) RMSNorm
  rmsnorm_kernel<<<Mrows, 256, 0, stream>>>(og, norm_w, onb);

  // 5) output projection -> f32 d_out
  gemm_bf16_kernel<0><<<ggrid, 256, 0, stream>>>(onb, wob, nullptr, (float*)d_out,
                                                 Mrows, Dd, Dd);
}
